// MultiHeadAttention_72696616452386
// MI455X (gfx1250) — matrix operand order repaired and checked by reading, not compiled, not run on hardware
//
#include <hip/hip_runtime.h>
#include <hip/hip_bf16.h>

// ---------------------------------------------------------------------------
// CDNA5 (gfx1250) wave32 WMMA multi-head attention forward.
//   pass 1: f32 -> bf16 casts (x, W_qkv, W_out)
//   pass 2: QKV GEMM  (bf16 WMMA, f32 acc, bias fused, TDM-staged A tiles)
//   pass 3: flash-style causal attention, one wave per 16-query block
//   pass 4: output projection GEMM -> f32 d_out (bias fused)
// ---------------------------------------------------------------------------

typedef __attribute__((ext_vector_type(16))) __bf16       v16bf;
typedef __attribute__((ext_vector_type(8)))  float        v8f;
typedef __attribute__((ext_vector_type(4)))  unsigned int u32x4;
typedef __attribute__((ext_vector_type(4)))  int          i32x4;
typedef __attribute__((ext_vector_type(8)))  int          i32x8;

union Frag {               // one WMMA 16-bit A/B operand: 16 bf16 = 32 bytes
    v16bf          v;
    u32x4          q[2];
    unsigned short s[16];
};

static __device__ __forceinline__ unsigned short f2bf(float f) {
    unsigned u = __float_as_uint(f);
    u += 0x7fffu + ((u >> 16) & 1u);     // round-to-nearest-even
    return (unsigned short)(u >> 16);
}

// ---------------------------------------------------------------------------
// Tensor Data Mover: async DMA of a 64(row) x 32(col) bf16 tile of a row-major
// [tensM, tensK] tensor into LDS.  D# per CDNA5 ISA.
// ---------------------------------------------------------------------------
#if __has_builtin(__builtin_amdgcn_tensor_load_to_lds)
#define HAVE_TDM 1
static __device__ __forceinline__ void tdm_load_tile_64x32(
    unsigned lds_off, const unsigned short* gptr, int tensK, int tensM)
{
    const unsigned long long ga = (unsigned long long)(uintptr_t)gptr;
    u32x4 g0;
    g0[0] = 1u;                                     // count=1, is_restore=0
    g0[1] = lds_off;                                // lds_addr (bytes)
    g0[2] = (unsigned)(ga & 0xffffffffull);         // global_addr[31:0]
    g0[3] = (unsigned)((ga >> 32) & 0x1ffffffull)   // global_addr[56:32]
          | (2u << 30);                             // type=2 ("image")
    i32x8 g1;
    g1[0] = 0x00010000;                             // data_size=1 (2 bytes)
    g1[1] = (int)(((unsigned)tensK & 0xffffu) << 16);            // dim0 lo
    g1[2] = (int)((((unsigned)tensK >> 16) & 0xffffu)
          | (((unsigned)tensM & 0xffffu) << 16));                // dim0 hi|dim1 lo
    g1[3] = (int)((((unsigned)tensM >> 16) & 0xffffu)
          | (32u << 16));                                        // dim1 hi|tile0=32
    g1[4] = 64;                                     // tile_dim1=64, tile_dim2=0
    g1[5] = tensK;                                  // tensor_dim0_stride lo
    g1[6] = 0;                                      // stride hi | dim1_stride lo
    g1[7] = 0;
    const i32x4 gz = {0, 0, 0, 0};
#if defined(__clang_major__) && __clang_major__ >= 23
    const i32x8 gz8 = {0, 0, 0, 0, 0, 0, 0, 0};
    __builtin_amdgcn_tensor_load_to_lds(g0, g1, gz, gz, gz8, 0);
#else
    __builtin_amdgcn_tensor_load_to_lds(g0, g1, gz, gz, 0);
#endif
}
#else
#define HAVE_TDM 0
#endif

// ---------------------------------------------------------------------------
__global__ void f32_to_bf16_kernel(const float* __restrict__ in,
                                   unsigned short* __restrict__ out, size_t n) {
    size_t i = (size_t)blockIdx.x * blockDim.x + threadIdx.x;
    if (i < n) out[i] = f2bf(in[i]);
}

// ---------------------------------------------------------------------------
// C[M,N] = A[M,K](bf16) * B[K,N](bf16) + bias (f32 acc).  Tile 64x64, BK=32,
// 4 waves / block.  A tiles double-buffered via async TDM (wave 0 issues,
// s_wait_tensorcnt + barrier publishes); B staged K-transposed so both WMMA
// operands are two 16B LDS reads each matching the 16-bit A/B VGPR layouts.
// All operand ds_loads are batched ahead of the 4-WMMA burst so the matrix
// pipe runs back-to-back instead of stalling on per-op s_wait_dscnt.
// ---------------------------------------------------------------------------
__global__ __launch_bounds__(128) void gemm_bf16_wmma(
    const unsigned short* __restrict__ A, const unsigned short* __restrict__ B,
    const float* __restrict__ bias, void* __restrict__ Cout,
    int M, int N, int K, int outBf16)
{
    __shared__ __align__(16) unsigned short As[2][64 * 32]; // [buf][row][k]
    __shared__ __align__(16) unsigned short Bst[64 * 32];   // [n][k] transposed

    const int tid  = threadIdx.x;
    const int lane = tid & 31;
    const int wave = tid >> 5;
    const int nBase = blockIdx.x * 64;
    const int mBase = blockIdx.y * 64;

    v8f acc[4] = {};

    const int bkrow = tid >> 2;          // 0..31
    const int bseg  = (tid & 3) * 16;    // 0/16/32/48 within BN

#if HAVE_TDM
    if (tid < 32)    // wave 0: kick off first A tile DMA
        tdm_load_tile_64x32((unsigned)(uintptr_t)&As[0][0],
                            A + (size_t)mBase * K, K, M);
#else
    const int arow  = tid >> 1;          // 0..63
    const int ahalf = (tid & 1) * 16;    // 0/16 within BK
#endif

    int buf = 0;
    for (int k0 = 0; k0 < K; k0 += 32) {
        __syncthreads();   // previous compute done: Bst / As[buf^1] reusable
        { // B tile 32x64 : coalesced b128 load, transposed scatter into Bst
            const unsigned short* src = B + (size_t)(k0 + bkrow) * N + nBase + bseg;
            Frag f;
            f.q[0] = *(const u32x4*)(src);
            f.q[1] = *(const u32x4*)(src + 8);
#pragma unroll
            for (int j = 0; j < 16; ++j)
                Bst[(bseg + j) * 32 + bkrow] = f.s[j];
            if (k0 + 32 < K)   // prefetch next B stripe -> global_prefetch_b8
                __builtin_prefetch(src + (size_t)32 * N, 0, 3);
        }
#if HAVE_TDM
        if (tid < 32) {
            if (k0 + 32 < K) {   // async-issue next A tile, then wait current
                tdm_load_tile_64x32((unsigned)(uintptr_t)&As[buf ^ 1][0],
                                    A + (size_t)mBase * K + (k0 + 32), K, M);
                __builtin_amdgcn_s_wait_tensorcnt((short)1);
            } else {
                __builtin_amdgcn_s_wait_tensorcnt((short)0);
            }
        }
#else
        { // fallback: synchronous A tile stage (64x32, 32B per thread)
            const unsigned short* src = A + (size_t)(mBase + arow) * K + k0 + ahalf;
            u32x4 q0 = *(const u32x4*)(src);
            u32x4 q1 = *(const u32x4*)(src + 8);
            *(u32x4*)(&As[0][arow * 32 + ahalf])     = q0;
            *(u32x4*)(&As[0][arow * 32 + ahalf + 8]) = q1;
        }
#endif
        __syncthreads();   // Bst + As[buf] visible to all waves

#if HAVE_TDM
        const unsigned short* asrc = &As[buf][0];
#else
        const unsigned short* asrc = &As[0][0];
#endif
        // A fragment (ISA 16-bit A layout: lanes<16 K0-7/K16-23, lanes>=16 +8)
        Frag af;
        const int arowf = wave * 16 + (lane & 15);
        const int koffA = (lane < 16) ? 0 : 8;
        af.q[0] = *(const u32x4*)(asrc + arowf * 32 + koffA);
        af.q[1] = *(const u32x4*)(asrc + arowf * 32 + koffA + 16);

        // batch all 4 B fragments (lanes<16 hold K0-7/K16-23, lanes>=16 +8)
        Frag bfr[4];
#pragma unroll
        for (int nt = 0; nt < 4; ++nt) {
            const int col   = nt * 16 + (lane & 15);
            const int koffB = (lane < 16) ? 0 : 8;
            bfr[nt].q[0] = *(const u32x4*)(&Bst[col * 32 + koffB]);
            bfr[nt].q[1] = *(const u32x4*)(&Bst[col * 32 + koffB + 16]);
        }
        // back-to-back WMMA burst (independent accumulators, shared A)
#pragma unroll
        for (int nt = 0; nt < 4; ++nt)
            acc[nt] = __builtin_amdgcn_wmma_f32_16x16x32_bf16(
                false, af.v, false, bfr[nt].v, (short)0, acc[nt], false, false);
        buf ^= 1;
    }

    // epilogue: C layout -> VGPR r holds row r (+8 for lanes 16-31), col=lane%16
    const int rowOff = wave * 16 + ((lane & 16) ? 8 : 0);
#pragma unroll
    for (int nt = 0; nt < 4; ++nt) {
        const int col = nBase + nt * 16 + (lane & 15);
        const float bv = bias[col];
#pragma unroll
        for (int r = 0; r < 8; ++r) {
            const size_t grow = (size_t)(mBase + rowOff + r);
            const float val = acc[nt][r] + bv;
            if (outBf16)
                ((unsigned short*)Cout)[grow * N + col] = f2bf(val);
            else
                ((float*)Cout)[grow * N + col] = val;
        }
    }
}

// ---------------------------------------------------------------------------
// Flash attention, causal.  One wave per (b, h, 16-query block).
// qkv buffer layout per token row (3072 bf16): head h -> [q:h*192, k:+64, v:+128]
// Output: attb[token][h*64 + d]  (bf16), softmax numerics in f32.
// ---------------------------------------------------------------------------
__global__ __launch_bounds__(32) void attn_wmma(
    const unsigned short* __restrict__ qkv, unsigned short* __restrict__ attout)
{
    __shared__ __align__(16) unsigned short p_lds[16 * 32]; // P tile [q][k]
    __shared__ __align__(16) unsigned short vt[64 * 32];    // V^T  [d][k]

    const int lane    = threadIdx.x;
    const int qb      = blockIdx.x;      // query block  0..127
    const int bh      = blockIdx.y;      // 0..31
    const int b       = bh >> 4;
    const int h       = bh & 15;
    const int S       = 2048;
    const size_t rs   = 3072;            // qkv row stride (elements)
    const size_t bTok = (size_t)b * S;
    const int qbase   = qb * 16;
    const int laneMod = lane & 15;
    const int hiHalf  = (lane >> 4) & 1;
    const int rowOff  = hiHalf ? 8 : 0;

    // Q A-fragments (hd=64 -> two K=32 chunks), loaded once
    Frag qa[2];
    {
        const unsigned short* qrow =
            qkv + (bTok + qbase + laneMod) * rs + (size_t)h * 192;
#pragma unroll
        for (int c = 0; c < 2; ++c) {
            const int d0 = c * 32 + (hiHalf ? 8 : 0);
            qa[c].q[0] = *(const u32x4*)(qrow + d0);
            qa[c].q[1] = *(const u32x4*)(qrow + d0 + 16);
        }
    }

    v8f o[4] = {};
    float mrun[8], lrun[8];
#pragma unroll
    for (int r = 0; r < 8; ++r) { mrun[r] = -1e30f; lrun[r] = 0.0f; }

    const int nsteps = (qbase + 15) / 32 + 1;   // causal: keys <= qbase+15
    for (int kb = 0; kb < nsteps; ++kb) {
        const int ktok0 = kb * 32;

        // ---- scores S = Q K^T for two 16-key tiles (loads batched) ----
        v8f sc[2];
#pragma unroll
        for (int t = 0; t < 2; ++t) {
            const unsigned short* krow =
                qkv + (bTok + ktok0 + t * 16 + laneMod) * rs + (size_t)h * 192 + 64;
            Frag kf[2];          // B frags: K0-7/K16-23 (+8 for lanes>=16)
#pragma unroll
            for (int c = 0; c < 2; ++c) {
                const int d0 = c * 32 + (hiHalf ? 8 : 0);
                kf[c].q[0] = *(const u32x4*)(krow + d0);
                kf[c].q[1] = *(const u32x4*)(krow + d0 + 16);
            }
            v8f a = {};
#pragma unroll
            for (int c = 0; c < 2; ++c)
                a = __builtin_amdgcn_wmma_f32_16x16x32_bf16(
                    false, qa[c].v, false, kf[c].v, (short)0, a, false, false);
            sc[t] = a;
        }

        // ---- stage V tile transposed into LDS: vt[d][k] ----
        {
            const unsigned short* vrow =
                qkv + (bTok + ktok0 + lane) * rs + (size_t)h * 192 + 128;
            union { u32x4 q[8]; unsigned short s[64]; } vr;
#pragma unroll
            for (int i = 0; i < 8; ++i) vr.q[i] = *(const u32x4*)(vrow + i * 8);
#pragma unroll
            for (int d = 0; d < 64; ++d) vt[d * 32 + lane] = vr.s[d];
        }

        // ---- online softmax (per-row m/l, 16-lane xor reductions) ----
        const float scale = 0.125f;              // 1/sqrt(64)
        float corrf[8];
#pragma unroll
        for (int r = 0; r < 8; ++r) {
            const int qg  = qbase + r + rowOff;  // global query index
            const int kc0 = ktok0 + laneMod;     // global key indices
            const int kc1 = kc0 + 16;
            float s0 = sc[0][r] * scale; if (kc0 > qg) s0 = -1e30f;
            float s1 = sc[1][r] * scale; if (kc1 > qg) s1 = -1e30f;
            float mx = fmaxf(s0, s1);
#pragma unroll
            for (int m = 1; m < 16; m <<= 1) mx = fmaxf(mx, __shfl_xor(mx, m, 32));
            const float mnew = fmaxf(mrun[r], mx);
            const float p0 = __expf(s0 - mnew);
            const float p1 = __expf(s1 - mnew);
            float rsum = p0 + p1;
#pragma unroll
            for (int m = 1; m < 16; m <<= 1) rsum += __shfl_xor(rsum, m, 32);
            const float corr = __expf(mrun[r] - mnew);
            lrun[r]  = lrun[r] * corr + rsum;
            mrun[r]  = mnew;
            corrf[r] = corr;
            // P (C layout) -> LDS for A-layout repack
            p_lds[(r + rowOff) * 32 + laneMod]      = f2bf(p0);
            p_lds[(r + rowOff) * 32 + laneMod + 16] = f2bf(p1);
        }
#pragma unroll
        for (int nt = 0; nt < 4; ++nt)
#pragma unroll
            for (int r = 0; r < 8; ++r) o[nt][r] *= corrf[r];

        __syncthreads();

        // ---- O += P @ V  (all operand ds_loads batched, then 4-WMMA burst) --
        Frag pf;
        {
            const int prow = laneMod;
            const int koff = hiHalf ? 8 : 0;
            pf.q[0] = *(const u32x4*)(&p_lds[prow * 32 + koff]);
            pf.q[1] = *(const u32x4*)(&p_lds[prow * 32 + koff + 16]);
        }
        Frag vf[4];
#pragma unroll
        for (int nt = 0; nt < 4; ++nt) {
            const int dim  = nt * 16 + laneMod;
            const int koff = hiHalf ? 8 : 0;
            vf[nt].q[0] = *(const u32x4*)(&vt[dim * 32 + koff]);
            vf[nt].q[1] = *(const u32x4*)(&vt[dim * 32 + koff + 16]);
        }
#pragma unroll
        for (int nt = 0; nt < 4; ++nt)
            o[nt] = __builtin_amdgcn_wmma_f32_16x16x32_bf16(
                false, pf.v, false, vf[nt].v, (short)0, o[nt], false, false);
        __syncthreads();
    }

    // ---- normalize and store (bf16) into [token][h*64 + d] ----
#pragma unroll
    for (int r = 0; r < 8; ++r) {
        const float inv  = 1.0f / lrun[r];
        const size_t tok = bTok + qbase + r + rowOff;
#pragma unroll
        for (int nt = 0; nt < 4; ++nt) {
            const int dcol = h * 64 + nt * 16 + laneMod;
            attout[tok * 1024 + dcol] = f2bf(o[nt][r] * inv);
        }
    }
}

// ---------------------------------------------------------------------------
extern "C" void kernel_launch(void* const* d_in, const int* in_sizes, int n_in,
                              void* d_out, int out_size, void* d_ws, size_t ws_size,
                              hipStream_t stream) {
    (void)in_sizes; (void)n_in; (void)out_size; (void)ws_size;

    const float* x     = (const float*)d_in[0];
    const float* W_qkv = (const float*)d_in[1];
    const float* b_qkv = (const float*)d_in[2];
    const float* W_out = (const float*)d_in[3];
    const float* b_out = (const float*)d_in[4];
    float* out = (float*)d_out;

    const int B = 2, S = 2048, D = 1024, H = 16;
    const int M  = B * S;      // 4096 tokens
    const int N3 = 3 * D;      // 3072

    // workspace layout (bf16 buffers), ~48 MiB total
    char* ws = (char*)d_ws;
    unsigned short* xb    = (unsigned short*)(ws);                        //  8 MiB
    unsigned short* wqkvb = (unsigned short*)(ws + ((size_t)8  << 20));   //  6 MiB
    unsigned short* woutb = (unsigned short*)(ws + ((size_t)14 << 20));   //  2 MiB
    unsigned short* qkvb  = (unsigned short*)(ws + ((size_t)16 << 20));   // 24 MiB
    unsigned short* attb  = (unsigned short*)(ws + ((size_t)40 << 20));   //  8 MiB

    const size_t nx = (size_t)M * D, nwq = (size_t)D * N3, nwo = (size_t)D * D;
    f32_to_bf16_kernel<<<(unsigned)((nx  + 255) / 256), 256, 0, stream>>>(x, xb, nx);
    f32_to_bf16_kernel<<<(unsigned)((nwq + 255) / 256), 256, 0, stream>>>(W_qkv, wqkvb, nwq);
    f32_to_bf16_kernel<<<(unsigned)((nwo + 255) / 256), 256, 0, stream>>>(W_out, woutb, nwo);

    // QKV projection: [4096,1024] x [1024,3072] -> bf16 qkv
    gemm_bf16_wmma<<<dim3(N3 / 64, M / 64), 128, 0, stream>>>(
        xb, wqkvb, b_qkv, qkvb, M, N3, D, /*outBf16=*/1);

    // causal flash attention: grid = (query blocks, B*H), 1 wave each
    attn_wmma<<<dim3(S / 16, B * H), 32, 0, stream>>>(qkvb, attb);

    // output projection: [4096,1024] x [1024,1024] -> f32 d_out
    gemm_bf16_wmma<<<dim3(D / 64, M / 64), 128, 0, stream>>>(
        attb, woutb, b_out, out, M, D, D, /*outBf16=*/0);
}
